// DilConv_40218073759941
// MI455X (gfx1250) — matrix operand order repaired and checked by reading, not compiled, not run on hardware
//
#include <hip/hip_runtime.h>

// ---------------------------------------------------------------------------
// Fused ReLU + depthwise dilated 3x3 conv + pointwise 1x1 conv (bf16 WMMA,
// fp32 accumulate) + training-mode BatchNorm for MI455X (gfx1250, wave32).
//
// Memory-bound (~19 GFLOP vs ~0.8 GB HBM traffic @ 23.3 TB/s). The conv
// pipeline is fused (depthwise result lives only in LDS, pre-swizzled into
// WMMA B-fragment layout) and the BN batch statistics are reduced directly
// from the WMMA accumulators so z is never re-read for stats.
// ---------------------------------------------------------------------------

typedef __attribute__((ext_vector_type(16))) __bf16 v16bf;
typedef __attribute__((ext_vector_type(8)))  float  v8f;

#define HW     64
#define CH     192
#define NB     64
#define PLANE  4096         // 64*64
#define KST    6            // 192 / 32 K-steps
#define NBLK   (NB * 64)    // 4096 conv blocks
#define NPIX   (NB * PLANE) // reduction length per channel

// ws layout (bytes):
//   [0,      1536)  : wsf    — mean[192], invstd[192]
//   [2048,   75776) : pwfrag — bf16 A-fragments (36864 elems)
//   [131072, ...)   : part1  — 4096*192 f32 block partial sums   (3 145 728 B)
//   [3276800,...)   : part2  — 4096*192 f32 block partial sumsq  (3 145 728 B)
#define WS_PART1_OFF 131072
#define WS_PART2_OFF 3276800
#define WS_NEEDED    (WS_PART2_OFF + (size_t)NBLK * CH * 4)

// ---- WMMA fragment index maps (ISA 7.12.2, wave32) -------------------------
// A (16x32 bf16, MxK): lanes 0-15 -> M=lane, lanes 16-31 -> M=lane-16.
//   element e (0..15): K = e + 16*hi   (hi = lane>>4), contiguous k order, same as B
// B (32x16 bf16, KxN): lanes 0-15 -> K=e, N=lane ; lanes 16-31 -> K=e+16
// C/D (16x16 f32):     reg r: M = r + 8*hi, N = lane&15
// ---------------------------------------------------------------------------

// Kernel 0: swizzle pw_w (fp32 192x192, [o][c]) into A-fragment bf16 layout.
__global__ __launch_bounds__(256) void pack_pw(const float* __restrict__ pw,
                                               __bf16* __restrict__ frag) {
  int gid  = blockIdx.x * 256 + threadIdx.x;   // 36864 total
  int f    = gid >> 9;
  int r    = gid & 511;
  int lane = r >> 4, e = r & 15;
  int mt   = f / KST, ks = f % KST;
  int lo   = lane & 15, hi = lane >> 4;
  int K    = e + 16 * hi;
  frag[gid] = (__bf16)pw[(mt * 16 + lo) * CH + ks * 32 + K];
}

// Kernel 1: per block = one (n, 16x4 pixel tile). relu + depthwise -> LDS
// B-fragments -> GEMM via v_wmma_f32_16x16x32_bf16 -> z, plus in-register
// per-block BN partial sums (if part1 != nullptr).
__global__ __launch_bounds__(256) void fused_dw_pw(
    const float* __restrict__ x, const float* __restrict__ dw,
    const __bf16* __restrict__ pwfrag, float* __restrict__ z,
    float* __restrict__ part1, float* __restrict__ part2) {
  __shared__ float dws[CH * 9];
  __shared__ __attribute__((aligned(32))) __bf16 ybf[4 * KST * 512]; // 24 KB
  __shared__ float p1s[2][CH], p2s[2][CH];                           // 3 KB

  const int tid  = threadIdx.x;
  const int blk  = blockIdx.x;
  const int n    = blk >> 6;
  const int tile = blk & 63;
  const int th   = (tile >> 2) * 4;   // row base (16 row-tiles of 4)
  const int tw   = (tile & 3) * 16;   // col base (4 col-tiles of 16)

  for (int i = tid; i < CH * 9; i += 256) dws[i] = dw[i];
  __syncthreads();

  // ---- depthwise stage: thread -> (pixel p = tid&63, channel group tid>>6)
  {
    const int p  = tid & 63;
    const int cg = tid >> 6;           // 4 groups x 48 channels
    const int hh = p >> 4, ww = p & 15;
    const int h  = th + hh, w = tw + ww;

    int  off[9];
    bool val[9];
#pragma unroll
    for (int i = 0; i < 3; ++i)
#pragma unroll
      for (int j = 0; j < 3; ++j) {
        int hy = h + (i - 1) * 2;      // dilation 2
        int wx = w + (j - 1) * 2;
        val[i * 3 + j] = (hy >= 0) && (hy < HW) && (wx >= 0) && (wx < HW);
        off[i * 3 + j] = hy * HW + wx;
      }

    const float* xb = x + (size_t)n * CH * PLANE;
    const int nt = p >> 4;
    const int lanelo = p & 15;

    for (int c = cg * 48; c < cg * 48 + 48; ++c) {
      const float* xp = xb + (size_t)c * PLANE;
      float acc = 0.f;
#pragma unroll
      for (int t = 0; t < 9; ++t) {
        if (val[t]) {
          float v = xp[off[t]];
          v = v > 0.f ? v : 0.f;       // ReLU before conv
          acc = fmaf(v, dws[c * 9 + t], acc);
        }
      }
      const int ks = c >> 5, cp = c & 31;
      const int hi = cp >> 4, e = cp & 15;
      ybf[(((nt * KST + ks) * 32) + hi * 16 + lanelo) * 16 + e] = (__bf16)acc;
    }
  }
  __syncthreads();

  // ---- GEMM stage: 8 waves; wave -> 3 M-tiles x 2 N-tiles
  const int lane = tid & 31;
  const int wv   = tid >> 5;
  const int mg   = wv & 3;             // 4 groups * 3 Mtiles = 12
  const int ng   = wv >> 2;            // 2 groups * 2 Ntiles = 4

  v8f acc[3][2] = {};
  for (int ks = 0; ks < KST; ++ks) {
    v16bf a[3], b[2];
#pragma unroll
    for (int i = 0; i < 3; ++i) {
      const int mt = mg * 3 + i;
      a[i] = *(const v16bf*)(pwfrag + ((size_t)(mt * KST + ks) * 32 + lane) * 16);
    }
#pragma unroll
    for (int j = 0; j < 2; ++j) {
      const int nt = ng * 2 + j;
      b[j] = *(const v16bf*)(ybf + ((nt * KST + ks) * 32 + lane) * 16);
    }
#pragma unroll
    for (int i = 0; i < 3; ++i)
#pragma unroll
      for (int j = 0; j < 2; ++j)
        acc[i][j] = __builtin_amdgcn_wmma_f32_16x16x32_bf16(
            false, a[i], false, b[j], (short)0, acc[i][j], false, false);
  }

  // ---- store z (NCHW), lanes 0-15 cover 16 consecutive w -> coalesced
  const int lo = lane & 15, hi = lane >> 4;
#pragma unroll
  for (int i = 0; i < 3; ++i) {
    const int mt = mg * 3 + i;
#pragma unroll
    for (int j = 0; j < 2; ++j) {
      const int nt = ng * 2 + j;
      const int h = th + nt, w = tw + lo;
      float* zp = z + (size_t)n * CH * PLANE + (size_t)(h * HW + w);
#pragma unroll
      for (int r = 0; r < 8; ++r)
        zp[(size_t)(mt * 16 + hi * 8 + r) * PLANE] = acc[i][j][r];
    }
  }

  // ---- fused BN partial stats straight from the accumulators
  if (part1 != nullptr) {
    float s1l[3][8], s2l[3][8];
#pragma unroll
    for (int i = 0; i < 3; ++i)
#pragma unroll
      for (int r = 0; r < 8; ++r) {
        const float z0 = acc[i][0][r], z1 = acc[i][1][r];
        s1l[i][r] = z0 + z1;
        s2l[i][r] = fmaf(z0, z0, z1 * z1);
      }
    // butterfly over the 16 N-lanes (stays inside each 16-lane half)
#pragma unroll
    for (int m = 1; m <= 8; m <<= 1)
#pragma unroll
      for (int i = 0; i < 3; ++i)
#pragma unroll
        for (int r = 0; r < 8; ++r) {
          s1l[i][r] += __shfl_xor(s1l[i][r], m, 32);
          s2l[i][r] += __shfl_xor(s2l[i][r], m, 32);
        }
    if (lo == 0) {
#pragma unroll
      for (int i = 0; i < 3; ++i)
#pragma unroll
        for (int r = 0; r < 8; ++r) {
          const int o = (mg * 3 + i) * 16 + hi * 8 + r;
          p1s[ng][o] = s1l[i][r];
          p2s[ng][o] = s2l[i][r];
        }
    }
    __syncthreads();
    if (tid < CH) {
      part1[(size_t)blk * CH + tid] = p1s[0][tid] + p1s[1][tid];
      part2[(size_t)blk * CH + tid] = p2s[0][tid] + p2s[1][tid];
    }
  }
}

// Kernel 2a: reduce per-block partials -> mean/invstd (deterministic order).
__global__ __launch_bounds__(256) void bn_stats_part(
    const float* __restrict__ part1, const float* __restrict__ part2,
    float* __restrict__ wsf) {
  const int o = blockIdx.x, tid = threadIdx.x;
  float s1 = 0.f, s2 = 0.f;
  for (int b = tid; b < NBLK; b += 256) {
    s1 += part1[(size_t)b * CH + o];
    s2 += part2[(size_t)b * CH + o];
  }
  __shared__ float sh1[256], sh2[256];
  sh1[tid] = s1; sh2[tid] = s2;
  __syncthreads();
  for (int s = 128; s > 0; s >>= 1) {
    if (tid < s) { sh1[tid] += sh1[tid + s]; sh2[tid] += sh2[tid + s]; }
    __syncthreads();
  }
  if (tid == 0) {
    const float inv = 1.f / (float)NPIX;
    const float m   = sh1[0] * inv;
    const float var = sh2[0] * inv - m * m;
    wsf[o]      = m;
    wsf[CH + o] = rsqrtf(var + 1e-5f);
  }
}

// Kernel 2b (fallback when ws too small): scan z for stats.
__global__ __launch_bounds__(256) void bn_stats(const float* __restrict__ z,
                                                float* __restrict__ wsf) {
  const int o = blockIdx.x, tid = threadIdx.x;
  float s1 = 0.f, s2 = 0.f;
  for (int idx = tid; idx < NPIX; idx += 256) {
    const float v = z[(((size_t)(idx >> 12) * CH + o) << 12) + (idx & (PLANE - 1))];
    s1 += v;
    s2 = fmaf(v, v, s2);
  }
  __shared__ float sh1[256], sh2[256];
  sh1[tid] = s1; sh2[tid] = s2;
  __syncthreads();
  for (int s = 128; s > 0; s >>= 1) {
    if (tid < s) { sh1[tid] += sh1[tid + s]; sh2[tid] += sh2[tid + s]; }
    __syncthreads();
  }
  if (tid == 0) {
    const float inv = 1.f / (float)NPIX;
    const float m   = sh1[0] * inv;
    const float var = sh2[0] * inv - m * m;
    wsf[o]      = m;
    wsf[CH + o] = rsqrtf(var + 1e-5f);
  }
}

// Kernel 3: in-place normalize. Each block covers 1024 consecutive floats,
// always inside a single (n,o) plane -> channel is block-uniform (SGPR math).
__global__ __launch_bounds__(256) void bn_apply(
    float* __restrict__ z, const float* __restrict__ wsf,
    const float* __restrict__ gamma, const float* __restrict__ beta) {
  const int o = (blockIdx.x >> 2) % CH;          // 4 blocks per 4096-plane
  const float scale = gamma[o] * wsf[CH + o];
  const float shift = fmaf(-wsf[o], scale, beta[o]);
  const size_t base = (size_t)blockIdx.x * 1024 + (size_t)threadIdx.x * 4;
  float4 v = *(const float4*)(z + base);
  v.x = fmaf(v.x, scale, shift);
  v.y = fmaf(v.y, scale, shift);
  v.z = fmaf(v.z, scale, shift);
  v.w = fmaf(v.w, scale, shift);
  *(float4*)(z + base) = v;
}

extern "C" void kernel_launch(void* const* d_in, const int* in_sizes, int n_in,
                              void* d_out, int out_size, void* d_ws, size_t ws_size,
                              hipStream_t stream) {
  const float* x     = (const float*)d_in[0];   // (64,192,64,64)
  const float* dw    = (const float*)d_in[1];   // (192,1,3,3)
  const float* pw    = (const float*)d_in[2];   // (192,192)
  const float* gamma = (const float*)d_in[3];   // (192)
  const float* beta  = (const float*)d_in[4];   // (192)
  float* z = (float*)d_out;                     // (64,192,64,64)

  float*  wsf    = (float*)d_ws;
  __bf16* pwfrag = (__bf16*)((char*)d_ws + 2048);
  const bool fused_stats = ws_size >= WS_NEEDED;
  float* part1 = fused_stats ? (float*)((char*)d_ws + WS_PART1_OFF) : nullptr;
  float* part2 = fused_stats ? (float*)((char*)d_ws + WS_PART2_OFF) : nullptr;

  pack_pw    <<<144,  256, 0, stream>>>(pw, pwfrag);
  fused_dw_pw<<<NBLK, 256, 0, stream>>>(x, dw, pwfrag, z, part1, part2);
  if (fused_stats)
    bn_stats_part<<<CH, 256, 0, stream>>>(part1, part2, wsf);
  else
    bn_stats<<<CH, 256, 0, stream>>>(z, wsf);
  bn_apply   <<<NB * CH * PLANE / 1024, 256, 0, stream>>>(z, wsf, gamma, beta);
}
